// BMNExtractor_37649683316958
// MI455X (gfx1250) — compile-verified, matrix operand order checked by reading, not run on hardware
//
#include <hip/hip_runtime.h>
#include <stdint.h>

// ---------------- types ----------------
typedef __bf16 bf16;
typedef __attribute__((ext_vector_type(16))) bf16  v16bf;
typedef __attribute__((ext_vector_type(8)))  bf16  v8bf;
typedef __attribute__((ext_vector_type(8)))  float v8f;
typedef __attribute__((ext_vector_type(4)))  unsigned int u32x4;
typedef __attribute__((ext_vector_type(8)))  int   i32x8;
typedef __attribute__((ext_vector_type(4)))  int   i32x4;

// Problem constants
#define TSC   128   // tscale (t / m)
#define DSC   128   // dscale (d)
#define NS    32    // num_sample (n)
#define CIN   256
#define CHID  128
#define CROI  512
#define COUT  128
#define BATCH 2
#define MT    8     // m-tile per workgroup -> 16 (b,m) columns

// ---------------- helpers ----------------
__device__ __forceinline__ v16bf cat8(v8bf lo, v8bf hi) {
  return __builtin_shufflevector(lo, hi, 0,1,2,3,4,5,6,7,8,9,10,11,12,13,14,15);
}
__device__ __forceinline__ v8f wmma_bf16(v16bf a, v16bf b, v8f c) {
  return __builtin_amdgcn_wmma_f32_16x16x32_bf16(false, a, false, b, (short)0, c, false, false);
}

// Issue one TDM tensor load: 3-D tile [t=64][n=32][m=8] fp32 from sample_mask
// (strides: m contiguous, n stride 16384 elems, t stride 524288 elems) -> LDS.
// This toolchain's builtin takes 6 operands: D# groups 0..3 plus an extra
// 8-dword group (uncharacterized; reserved-zero per D# conventions) and cpol.
__device__ __forceinline__ void tdm_issue_mask(uint64_t gaddr, unsigned ldsoff) {
  u32x4 g0;
  g0[0] = 1u;                                       // count=1, user descriptor
  g0[1] = ldsoff;                                   // lds_addr (bytes)
  g0[2] = (unsigned)(gaddr & 0xffffffffu);          // global_addr[31:0]
  g0[3] = (unsigned)((gaddr >> 32) & 0x1ffffffu)    // global_addr[56:32]
        | 0x80000000u;                              // type=2 ("image")
  i32x8 g1;
  g1[0] = (int)(2u << 16);                          // data_size = 4B
  g1[1] = (int)((unsigned)MT << 16);                // tensor_dim0 = 8
  g1[2] = (int)(32u << 16);                         // tensor_dim1 = 32
  g1[3] = (int)((unsigned)MT << 16);                // tile_dim0 = 8
  g1[4] = (int)(32u | (64u << 16));                 // tile_dim1=32, tile_dim2=64
  g1[5] = 16384;                                    // tensor_dim0_stride = 16384
  g1[6] = 0;                                        // stride0 hi / stride1 lo16 (0x80000 low16 = 0)
  g1[7] = 8;                                        // tensor_dim1_stride = 0x80000 >> 16
  i32x4 g2;
  g2[0] = 64;                                       // tensor_dim2 = 64 (per half)
  g2[1] = 1;                                        // tensor_dim3 = 1 (defensive, tile_dim3 unused)
  g2[2] = 0;                                        // tensor_dim2_stride lo (unused)
  g2[3] = 0;                                        // tile_dim3 = 0 (unused)
  i32x4 g3;
  g3[0] = 0;
  g3[1] = (int)(1u << 16);                          // tensor_dim4 = 1 (defensive)
  g3[2] = 0;
  g3[3] = 0;
  i32x8 g4;                                         // extra group (reserved) = 0
  g4[0] = 0; g4[1] = 0; g4[2] = 0; g4[3] = 0;
  g4[4] = 0; g4[5] = 0; g4[6] = 0; g4[7] = 0;
  __builtin_amdgcn_tensor_load_to_lds(g0, g1, g2, g3, g4, 0);
}

// ---------------- prep kernels ----------------
// Conv1d(256->128, k=3, SAME) + ReLU, output bf16 h[(b*128+co)*128 + t]
__global__ __launch_bounds__(128) void conv_relu_kernel(
    const float* __restrict__ x, const float* __restrict__ wr,
    const float* __restrict__ br, bf16* __restrict__ hsg)
{
  const int t  = threadIdx.x;
  const int bc = blockIdx.x;           // b*128 + co
  const int b  = bc >> 7, co = bc & 127;
  float acc = br[co];
  const float* xb = x  + (size_t)b  * CIN * TSC;
  const float* wc = wr + (size_t)co * CIN * 3;
  for (int ci = 0; ci < CIN; ++ci) {
    const float* xr = xb + ci * TSC;
    const float* wk = wc + ci * 3;
    float xm1 = (t > 0)       ? xr[t - 1] : 0.f;
    float x0  = xr[t];
    float xp1 = (t < TSC - 1) ? xr[t + 1] : 0.f;
    acc += xm1 * wk[0] + x0 * wk[1] + xp1 * wk[2];
  }
  acc = acc > 0.f ? acc : 0.f;
  hsg[(size_t)bc * TSC + t] = (bf16)acc;
}

// fp32 -> bf16 copies of w3d [512 x 4096] and w2d [128 x 512]
__global__ __launch_bounds__(256) void cvt_weights_kernel(
    const float* __restrict__ w3d, const float* __restrict__ w2d,
    bf16* __restrict__ w3b, bf16* __restrict__ w2b)
{
  const int n3 = CROI * CHID * NS;     // 2097152
  const int n2 = COUT * CROI;          // 65536
  for (int i = blockIdx.x * blockDim.x + threadIdx.x; i < n3 + n2;
       i += gridDim.x * blockDim.x) {
    if (i < n3) w3b[i] = (bf16)w3d[i];
    else        w2b[i - n3] = (bf16)w2d[i - n3];
  }
}

// ---------------- fused BMN kernel ----------------
// grid (128 d, 16 m-tiles), block 256 (8 wave32).
// Fragment index math follows ISA 7.12.2 (wave32, 16-bit A/B, f32 C/D):
//   A/B lane l: row/col = l&15, kbase = (l>>4)*8; elem e -> K = ((e>>3)<<4)+kbase+(e&7)
//   D lane l, vgpr r:  N = l&15, M = r + 8*(l>>4)
__global__ __launch_bounds__(256) void bmn_main(
    const bf16* __restrict__ hsg, const bf16* __restrict__ w3b,
    const bf16* __restrict__ w2b, const float* __restrict__ b3,
    const float* __restrict__ b2, const float* __restrict__ mask,
    float* __restrict__ out)
{
  __shared__ __align__(16) float staging[MT * NS * 64];      // 64 KB TDM landing pad
  __shared__ __align__(32) bf16  hs[BATCH * CHID * TSC];     // 64 KB h (both batches)
  __shared__ __align__(32) bf16  Bbuf[MT * 2 * 4 * 512];     // 64 KB mask B-frags
  __shared__ __align__(32) bf16  Pbuf[16 * 512];             // 16 KB P  B-frags (per c-tile)
  __shared__ __align__(32) bf16  m3buf[16 * 512];            // 16 KB m3 B-frags

  const int d    = blockIdx.x;
  const int m0   = blockIdx.y * MT;
  const int tid  = threadIdx.x;
  const int lane = tid & 31;
  const int w    = tid >> 5;

  const uint64_t gaddr0 =
      (uint64_t)(uintptr_t)(mask + (size_t)d * TSC + m0);
  const unsigned ldsoff = (unsigned)(uintptr_t)&staging[0];

  if (w == 0) tdm_issue_mask(gaddr0, ldsoff);                // half 0: t in [0,64)

  // cooperative copy of h (bf16, 64KB) into LDS while TDM is in flight
  {
    const uint4* src = (const uint4*)hsg;
    uint4*       dst = (uint4*)hs;
    for (int i = tid; i < (BATCH * CHID * TSC * 2) / 16; i += 256) dst[i] = src[i];
  }

  // stage mask halves: TDM -> staging -> bf16 B-fragments in Bbuf
  for (int h = 0; h < 2; ++h) {
    if (w == 0) __builtin_amdgcn_s_wait_tensorcnt(0);
    __syncthreads();                                          // staging ready everywhere
    for (int i = tid; i < 32768; i += 256) {
      const int e   = i & 15;
      const int l   = (i >> 4) & 31;
      const int ksl = (i >> 9) & 1;
      const int nt  = (i >> 10) & 1;
      const int m   = i >> 11;
      const int ks  = 2 * h + ksl;
      const int tl  = ksl * 32 + ((e >> 3) << 4) + ((l >> 4) << 3) + (e & 7);
      const int n   = nt * 16 + (l & 15);
      Bbuf[(((m * 2 + nt) * 4 + ks) * 32 + l) * 16 + e] =
          (bf16)staging[(tl * NS + n) * MT + m];
    }
    __syncthreads();                                          // staging free again
    if (h == 0 && w == 0)
      tdm_issue_mask(gaddr0 + (uint64_t)64 * (NS * DSC * TSC) * 4, ldsoff); // t in [64,128)
  }

  // GEMM1 (h x mask -> P) fused with GEMM2 (w3d x P), K2 = c*32+n
  v8f accD2[4] = {};                     // this wave's o-rows: [64w, 64w+64)
  const int b_g1 = w >> 2;               // batch handled by this wave in GEMM1

  for (int ct = 0; ct < 8; ++ct) {       // c-tiles of 16
    v8f acc1[4] = {};
#pragma unroll
    for (int ks = 0; ks < 4; ++ks) {     // K1 = t, 4 steps of 32
      const int abase = ((b_g1 * CHID + ct * 16 + (lane & 15)) * TSC)
                      + ks * 32 + ((lane >> 4) << 3);
      const v8bf alo = *(const v8bf*)&hs[abase];
      const v8bf ahi = *(const v8bf*)&hs[abase + 16];
      const v16bf a  = cat8(alo, ahi);
#pragma unroll
      for (int i = 0; i < 4; ++i) {      // 4 (m,nt) tiles per wave
        const int q  = w * 4 + i;
        const int m  = (q >> 1) & 7, nt = q & 1;
        const v16bf bb =
            *(const v16bf*)&Bbuf[(((m * 2 + nt) * 4 + ks) * 32 + lane) * 16];
        acc1[i] = wmma_bf16(a, bb, acc1[i]);
      }
    }
    __syncthreads();                     // previous GEMM2 done reading Pbuf
    // scatter P tiles (bf16) into GEMM2 B-fragment layout: kstep = c_local, K = n
#pragma unroll
    for (int i = 0; i < 4; ++i) {
      const int q    = w * 4 + i;
      const int m    = (q >> 1) & 7, nt = q & 1;
      const int colP = b_g1 * MT + m;
      const int n    = nt * 16 + (lane & 15);
      const int l2   = colP + (((n >> 3) & 1) << 4);
      const int e2   = ((n >> 4) << 3) + (n & 7);
      const int kkb  = (lane >> 4) << 3;
#pragma unroll
      for (int r = 0; r < 8; ++r)
        Pbuf[((r + kkb) * 32 + l2) * 16 + e2] = (bf16)acc1[i][r];
    }
    __syncthreads();                     // Pbuf complete
    // GEMM2 partial: A = w3d_bf16 [512 x 4096] (global, layout-native)
    for (int kkl = 0; kkl < 16; ++kkl) {
      const v16bf pb = *(const v16bf*)&Pbuf[(kkl * 32 + lane) * 16];
      const int kkg = ct * 16 + kkl;
#pragma unroll
      for (int i = 0; i < 4; ++i) {
        const int o = (w * 4 + i) * 16 + (lane & 15);
        const bf16* ap = &w3b[(size_t)o * (CHID * NS) + kkg * 32 + ((lane >> 4) << 3)];
        accD2[i] = wmma_bf16(cat8(*(const v8bf*)ap, *(const v8bf*)(ap + 16)),
                             pb, accD2[i]);
      }
    }
  }

  // m3 = relu(D2 + b3d) -> bf16 B-frags for GEMM3 (K3 = o, 16 steps of 32)
  const int col = lane & 15;
#pragma unroll
  for (int i = 0; i < 4; ++i) {
#pragma unroll
    for (int r = 0; r < 8; ++r) {
      const int o = (w * 4 + i) * 16 + r + ((lane >> 4) << 3);
      float v = accD2[i][r] + b3[o];
      v = v > 0.f ? v : 0.f;
      const int n  = o & 31;
      const int l2 = col + (((n >> 3) & 1) << 4);
      const int e2 = ((n >> 4) << 3) + (n & 7);
      m3buf[((o >> 5) * 32 + l2) * 16 + e2] = (bf16)v;
    }
  }
  __syncthreads();

  // GEMM3: each wave owns one 16-row j-tile of w2d [128 x 512]
  v8f acc3 = {};
  for (int kk2 = 0; kk2 < 16; ++kk2) {
    const v16bf mb = *(const v16bf*)&m3buf[(kk2 * 32 + lane) * 16];
    const int j = w * 16 + (lane & 15);
    const bf16* ap = &w2b[(size_t)j * CROI + kk2 * 32 + ((lane >> 4) << 3)];
    acc3 = wmma_bf16(cat8(*(const v8bf*)ap, *(const v8bf*)(ap + 16)), mb, acc3);
  }

  // out = relu(D3 + b2d), scatter to [B][COUT][D][M]
  const int bb2 = col >> 3, ml = col & 7;
#pragma unroll
  for (int r = 0; r < 8; ++r) {
    const int j = w * 16 + r + ((lane >> 4) << 3);
    float v = acc3[r] + b2[j];
    v = v > 0.f ? v : 0.f;
    out[(((size_t)bb2 * COUT + j) * DSC + d) * TSC + (m0 + ml)] = v;
  }
}

// ---------------- launcher ----------------
extern "C" void kernel_launch(void* const* d_in, const int* in_sizes, int n_in,
                              void* d_out, int out_size, void* d_ws, size_t ws_size,
                              hipStream_t stream)
{
  const float* x    = (const float*)d_in[0];
  const float* wred = (const float*)d_in[1];
  const float* bred = (const float*)d_in[2];
  const float* w3d  = (const float*)d_in[3];
  const float* b3d  = (const float*)d_in[4];
  const float* w2d  = (const float*)d_in[5];
  const float* b2d  = (const float*)d_in[6];
  const float* mask = (const float*)d_in[7];
  float* out = (float*)d_out;

  char* ws = (char*)d_ws;
  bf16* hsg = (bf16*)ws;                            // 64 KB
  bf16* w3b = (bf16*)(ws + 65536);                  // 4 MiB
  bf16* w2b = (bf16*)(ws + 65536 + 4194304);        // 128 KB

  conv_relu_kernel<<<dim3(BATCH * CHID), dim3(TSC), 0, stream>>>(x, wred, bred, hsg);
  cvt_weights_kernel<<<dim3(2048), dim3(256), 0, stream>>>(w3d, w2d, w3b, w2b);
  bmn_main<<<dim3(DSC, TSC / MT), dim3(256), 0, stream>>>(hsg, w3b, w2b, b3d, b2d, mask, out);
}
